// MultiHeadSelfAttention_59914793779398
// MI455X (gfx1250) — compile-verified, matrix operand order checked by reading, not run on hardware
//
#include <hip/hip_runtime.h>
#include <hip/hip_bf16.h>

#define D_MODEL 1024
#define NUM_HEADS 16
#define HEAD_DIM 64
#define BATCH 4
#define SEQ 2048
#define M_TOK (BATCH * SEQ)

typedef __attribute__((ext_vector_type(16))) _Float16 v16h;
typedef __attribute__((ext_vector_type(8)))  _Float16 v8h;
typedef __attribute__((ext_vector_type(8)))  float    v8f;
typedef int i32x4 __attribute__((vector_size(16)));

#if __has_builtin(__builtin_amdgcn_global_load_async_to_lds_b128)
#define USE_ASYNC_LDS 1
#else
#define USE_ASYNC_LDS 0
#endif

__device__ __forceinline__ v8f zero_v8f() {
  v8f z = {0.f, 0.f, 0.f, 0.f, 0.f, 0.f, 0.f, 0.f};
  return z;
}

// Build a 16-half WMMA fragment for this lane from an LDS row.
// Per CDNA5 ISA (16-bit A 16x32 / B 32x16 layouts): lane (l, hi) needs the
// two contiguous 8-half chunks at K offsets [ks + hi*8, +8) and
// [ks + 16 + hi*8, +8).  rowbase must be 16B aligned.
__device__ __forceinline__ v16h load_frag(const _Float16* rowbase, int hi, int ks) {
  v8h lo = *(const v8h*)(rowbase + ks + hi * 8);
  v8h hh = *(const v8h*)(rowbase + ks + 16 + hi * 8);
  return __builtin_shufflevector(lo, hh, 0, 1, 2, 3, 4, 5, 6, 7,
                                 8, 9, 10, 11, 12, 13, 14, 15);
}

__device__ __forceinline__ v8f wmma_f16(v16h a, v16h b, v8f c) {
  return __builtin_amdgcn_wmma_f32_16x16x32_f16(false, a, false, b, (short)0, c,
                                                false, false);
}

#if USE_ASYNC_LDS
// 16B async global->LDS copy (gfx1250 GLOBAL_LOAD_ASYNC_TO_LDS_B128,
// tracked by ASYNCcnt).  Builtin takes (i32x4 AS1*, i32x4 AS3*, imm, imm).
__device__ __forceinline__ void async_cp16(const _Float16* g, _Float16* l) {
  __builtin_amdgcn_global_load_async_to_lds_b128(
      (__attribute__((address_space(1))) i32x4*)g,
      (__attribute__((address_space(3))) i32x4*)l, 0, 0);
}
#endif

// Copy one 64B chunk (4 x 16B) global -> LDS, async when available.
__device__ __forceinline__ void stage64(const _Float16* g, _Float16* l) {
#if USE_ASYNC_LDS
  async_cp16(g + 0, l + 0);
  async_cp16(g + 8, l + 8);
  async_cp16(g + 16, l + 16);
  async_cp16(g + 24, l + 24);
#else
  const v8h* s = (const v8h*)g;
  v8h* d = (v8h*)l;
  d[0] = s[0]; d[1] = s[1]; d[2] = s[2]; d[3] = s[3];
#endif
}

// ---------------------------------------------------------------------------
// fp32 -> f16 conversion (weights)
// ---------------------------------------------------------------------------
__global__ void cvt_f32_to_f16(const float* __restrict__ src,
                               _Float16* __restrict__ dst, int n) {
  int i = blockIdx.x * blockDim.x + threadIdx.x;
  int stride = gridDim.x * blockDim.x;
  for (; i < n; i += stride) dst[i] = (_Float16)src[i];
}

// ---------------------------------------------------------------------------
// GEMM: Out = A(M x 1024) @ W(1024 x 1024) + bias
// MODE 0: A fp32, Out f16 in [B,H,S,hd] layout        (Q, K projections)
// MODE 1: A fp32, Out f16 in [B,H,hd,S] layout (V^T)  (V projection)
// MODE 2: A f16,  Out fp32 row-major [M,1024]         (output projection)
// Block: 256 thr = 8 waves as 4(M) x 2(N). Block tile 128x128, wave 32x64.
// Double-buffered LDS + register prefetch of the next k-block.
// ---------------------------------------------------------------------------
template <int MODE>
__global__ __launch_bounds__(256) void gemm_kernel(
    const void* __restrict__ Ap, const _Float16* __restrict__ W,
    const float* __restrict__ bias, void* __restrict__ Outp) {
  __shared__ alignas(16) _Float16 As[2][128][40];  // [m][k], padded stride
  __shared__ alignas(16) _Float16 Bs[2][128][40];  // [n][k] (W tile transposed)

  const int tid = threadIdx.x;
  const int bm = blockIdx.y * 128;
  const int bn = blockIdx.x * 128;
  const int wave = tid >> 5, lane = tid & 31;
  const int wm = wave & 3, wn = wave >> 2;
  const int l = lane & 15, hi = lane >> 4;

  v8f acc[2][4];
#pragma unroll
  for (int i = 0; i < 2; i++)
#pragma unroll
    for (int j = 0; j < 4; j++) acc[i][j] = zero_v8f();

  const int arow = tid >> 1, ac0 = (tid & 1) * 16;   // A: 128 rows x 32 cols
  const int brow = tid >> 3, bn0 = (tid & 7) * 16;   // B: 32 rows x 128 cols

  // register staging for the next k-block
  v8h a0, a1, b0, b1;

  auto load_block = [&](int kb) {
    if (MODE < 2) {
      const float* Ag = (const float*)Ap + (size_t)(bm + arow) * D_MODEL + kb + ac0;
      float4 x0 = ((const float4*)Ag)[0];
      float4 x1 = ((const float4*)Ag)[1];
      float4 x2 = ((const float4*)Ag)[2];
      float4 x3 = ((const float4*)Ag)[3];
      a0 = v8h{(_Float16)x0.x, (_Float16)x0.y, (_Float16)x0.z, (_Float16)x0.w,
               (_Float16)x1.x, (_Float16)x1.y, (_Float16)x1.z, (_Float16)x1.w};
      a1 = v8h{(_Float16)x2.x, (_Float16)x2.y, (_Float16)x2.z, (_Float16)x2.w,
               (_Float16)x3.x, (_Float16)x3.y, (_Float16)x3.z, (_Float16)x3.w};
      if (kb + 32 < D_MODEL) __builtin_prefetch(Ag + 32, 0, 1);
    } else {
      const _Float16* Ag =
          (const _Float16*)Ap + (size_t)(bm + arow) * D_MODEL + kb + ac0;
      a0 = ((const v8h*)Ag)[0];
      a1 = ((const v8h*)Ag)[1];
      if (kb + 32 < D_MODEL) __builtin_prefetch(Ag + 32, 0, 1);
    }
    const _Float16* Wg = W + (size_t)(kb + brow) * D_MODEL + bn + bn0;
    b0 = ((const v8h*)Wg)[0];
    b1 = ((const v8h*)Wg)[1];
  };

  load_block(0);
  int p = 0;
  for (int kb = 0; kb < D_MODEL; kb += 32) {
    // store staged regs into buffer p
    *(v8h*)&As[p][arow][ac0] = a0;
    *(v8h*)&As[p][arow][ac0 + 8] = a1;
#pragma unroll
    for (int j = 0; j < 8; j++) Bs[p][bn0 + j][brow] = b0[j];
#pragma unroll
    for (int j = 0; j < 8; j++) Bs[p][bn0 + 8 + j][brow] = b1[j];
    if (kb + 32 < D_MODEL) load_block(kb + 32);  // overlaps with compute
    __syncthreads();

    v16h af[2], bf[4];
#pragma unroll
    for (int mi = 0; mi < 2; mi++)
      af[mi] = load_frag(&As[p][wm * 32 + mi * 16 + l][0], hi, 0);
#pragma unroll
    for (int ni = 0; ni < 4; ni++)
      bf[ni] = load_frag(&Bs[p][wn * 64 + ni * 16 + l][0], hi, 0);
#pragma unroll
    for (int mi = 0; mi < 2; mi++)
#pragma unroll
      for (int ni = 0; ni < 4; ni++)
        acc[mi][ni] = wmma_f16(af[mi], bf[ni], acc[mi][ni]);
    p ^= 1;
  }

  // ---- epilogue: C/D layout -> lane holds rows hi*8+r, col l ----
#pragma unroll
  for (int mi = 0; mi < 2; mi++) {
#pragma unroll
    for (int ni = 0; ni < 4; ni++) {
      const int gm = bm + wm * 32 + mi * 16 + hi * 8;
      const int gn = bn + wn * 64 + ni * 16 + l;
      const float bv = bias[gn];
      if (MODE == 0) {
        _Float16* Out = (_Float16*)Outp;
        const int h = gn >> 6, d = gn & 63;
#pragma unroll
        for (int r = 0; r < 8; r++) {
          const int row = gm + r, b = row >> 11, s = row & (SEQ - 1);
          Out[((((size_t)b * NUM_HEADS + h) * SEQ + s) << 6) + d] =
              (_Float16)(acc[mi][ni][r] + bv);
        }
      } else if (MODE == 1) {
        _Float16* Out = (_Float16*)Outp;
        const int h = gn >> 6, d = gn & 63;
        const int b = gm >> 11, s = gm & (SEQ - 1);
        v8h o;
#pragma unroll
        for (int r = 0; r < 8; r++) o[r] = (_Float16)(acc[mi][ni][r] + bv);
        *(v8h*)(Out + ((((size_t)b * NUM_HEADS + h) * HEAD_DIM + d) << 11) + s) = o;
      } else {
        float* Out = (float*)Outp;
#pragma unroll
        for (int r = 0; r < 8; r++)
          Out[(size_t)(gm + r) * D_MODEL + gn] = acc[mi][ni][r] + bv;
      }
    }
  }
}

// ---------------------------------------------------------------------------
// Flash attention: grid (S/64, B*H), 128 threads = 4 waves x 16 q-rows.
// Qh, Kh: f16 [B,H,S,hd];  Vt: f16 [B,H,hd,S];  O: f16 [B,S,D]
// K/V tiles double-buffered; staged with async global->LDS when available.
// Q is pre-scaled by 1/sqrt(hd)=0.125 (exact in f16) and its fragments are
// hoisted out of the kv loop.
// ---------------------------------------------------------------------------
__global__ __launch_bounds__(128) void attn_kernel(
    const _Float16* __restrict__ Qh, const _Float16* __restrict__ Kh,
    const _Float16* __restrict__ Vt, _Float16* __restrict__ O) {
  __shared__ alignas(16) _Float16 Qs[64][72];         // [q][d]
  __shared__ alignas(16) _Float16 Ks[2][64][72];      // [kv][d]
  __shared__ alignas(16) _Float16 Vs[2][64][72];      // [d][kv]
  __shared__ alignas(16) _Float16 Ps[4][16][72];      // per-wave P tile [q][kv]

  const int tid = threadIdx.x;
  const int bh = blockIdx.y;           // b*16 + h
  const int q0 = blockIdx.x * 64;
  const int wave = tid >> 5, lane = tid & 31;
  const int l = lane & 15, hi = lane >> 4;
  const int srow = tid >> 1, sc0 = (tid & 1) * 32;  // cooperative staging

  const _Float16* Kg0 = Kh + (size_t)bh * SEQ * HEAD_DIM;
  const _Float16* Vg0 = Vt + (size_t)bh * HEAD_DIM * SEQ;

  // issue tile 0 K/V staging (async overlaps the Q staging below)
  stage64(Kg0 + (size_t)srow * HEAD_DIM + sc0, &Ks[0][srow][sc0]);
  stage64(Vg0 + (size_t)srow * SEQ + sc0, &Vs[0][srow][sc0]);

  // stage Q tile once, pre-scaled by 1/sqrt(HEAD_DIM) (exact power of two)
  {
    const _Float16* Qg =
        Qh + ((size_t)bh * SEQ + q0) * HEAD_DIM + (size_t)srow * HEAD_DIM + sc0;
    const v8h* src = (const v8h*)Qg;
    v8h* dst = (v8h*)&Qs[srow][sc0];
#pragma unroll
    for (int j = 0; j < 4; j++) {
      v8h x = src[j];
#pragma unroll
      for (int e = 0; e < 8; e++) x[e] = x[e] * (_Float16)0.125f;
      dst[j] = x;
    }
  }
  __syncthreads();

  // loop-invariant Q fragments
  v16h qfrag[2];
  qfrag[0] = load_frag(&Qs[wave * 16 + l][0], hi, 0);
  qfrag[1] = load_frag(&Qs[wave * 16 + l][0], hi, 32);

  float m_prev[8], lsum[8];
#pragma unroll
  for (int r = 0; r < 8; r++) { m_prev[r] = -3.0e38f; lsum[r] = 0.f; }
  v8f acc[4];
#pragma unroll
  for (int n = 0; n < 4; n++) acc[n] = zero_v8f();

  int p = 0;
  for (int kt = 0; kt < SEQ; kt += 64) {
#if USE_ASYNC_LDS
    asm volatile("s_wait_asynccnt 0" ::: "memory");
#endif
    __syncthreads();  // buffer p fully staged, previous reads of 1-p done

    if (kt + 64 < SEQ) {  // stage next tile into 1-p, overlapping compute
      stage64(Kg0 + (size_t)(kt + 64 + srow) * HEAD_DIM + sc0, &Ks[1 - p][srow][sc0]);
      stage64(Vg0 + (size_t)srow * SEQ + kt + 64 + sc0, &Vs[1 - p][srow][sc0]);
    }

    // scores: 16(q) x 64(kv) per wave, K-dim = 64 (two 32-steps)
    v8f sc[4];
#pragma unroll
    for (int n = 0; n < 4; n++) sc[n] = zero_v8f();
#pragma unroll
    for (int ksi = 0; ksi < 2; ksi++) {
#pragma unroll
      for (int n = 0; n < 4; n++) {
        v16h bk = load_frag(&Ks[p][n * 16 + l][0], hi, ksi * 32);
        sc[n] = wmma_f16(qfrag[ksi], bk, sc[n]);
      }
    }

    // online softmax; rows live on 16 lanes of each half-wave
    float sf[8];
#pragma unroll
    for (int r = 0; r < 8; r++) {
      float v = fmaxf(fmaxf(sc[0][r], sc[1][r]), fmaxf(sc[2][r], sc[3][r]));
#pragma unroll
      for (int m = 1; m < 16; m <<= 1) v = fmaxf(v, __shfl_xor(v, m, 32));
      const float mn = fmaxf(m_prev[r], v);
      sf[r] = __expf(m_prev[r] - mn);
      m_prev[r] = mn;
    }
    float rs[8];
#pragma unroll
    for (int r = 0; r < 8; r++) rs[r] = 0.f;
#pragma unroll
    for (int n = 0; n < 4; n++) {
#pragma unroll
      for (int r = 0; r < 8; r++) {
        const float pv = __expf(sc[n][r] - m_prev[r]);
        rs[r] += pv;
        Ps[wave][hi * 8 + r][n * 16 + l] = (_Float16)pv;
      }
    }
#pragma unroll
    for (int r = 0; r < 8; r++) {
      float v = rs[r];
#pragma unroll
      for (int m = 1; m < 16; m <<= 1) v += __shfl_xor(v, m, 32);
      lsum[r] = lsum[r] * sf[r] + v;
    }
#pragma unroll
    for (int n = 0; n < 4; n++)
#pragma unroll
      for (int r = 0; r < 8; r++) acc[n][r] *= sf[r];

    // same-wave LDS is in order; explicit wait before re-reading P
    asm volatile("s_wait_dscnt 0" ::: "memory");

    // acc += P @ V  (K-dim = 64 kv, two 32-steps)
#pragma unroll
    for (int ksi = 0; ksi < 2; ksi++) {
      v16h pa = load_frag(&Ps[wave][l][0], hi, ksi * 32);
#pragma unroll
      for (int n = 0; n < 4; n++) {
        v16h vb = load_frag(&Vs[p][n * 16 + l][0], hi, ksi * 32);
        acc[n] = wmma_f16(pa, vb, acc[n]);
      }
    }
    p ^= 1;
  }

  // epilogue: O[b, s, h*64 + d] f16
  const int b = bh >> 4, h = bh & 15;
#pragma unroll
  for (int r = 0; r < 8; r++) {
    const float inv = 1.0f / lsum[r];
    const int tok = q0 + wave * 16 + hi * 8 + r;
    const size_t base = ((size_t)b * SEQ + tok) * D_MODEL + h * HEAD_DIM;
#pragma unroll
    for (int n = 0; n < 4; n++)
      O[base + n * 16 + l] = (_Float16)(acc[n][r] * inv);
  }
}

// ---------------------------------------------------------------------------
extern "C" void kernel_launch(void* const* d_in, const int* in_sizes, int n_in,
                              void* d_out, int out_size, void* d_ws,
                              size_t ws_size, hipStream_t stream) {
  const float* query = (const float*)d_in[0];
  const float* key   = (const float*)d_in[1];
  const float* value = (const float*)d_in[2];
  const float* Wq = (const float*)d_in[3];
  const float* bq = (const float*)d_in[4];
  const float* Wk = (const float*)d_in[5];
  const float* bk = (const float*)d_in[6];
  const float* Wv = (const float*)d_in[7];
  const float* bv = (const float*)d_in[8];
  const float* Wo = (const float*)d_in[9];
  const float* bo = (const float*)d_in[10];
  float* out = (float*)d_out;

  const size_t WSZ = (size_t)D_MODEL * D_MODEL;  // 1M elems
  const size_t TSZ = (size_t)M_TOK * D_MODEL;    // 8M elems
  char* ws = (char*)d_ws;
  _Float16* Wq16 = (_Float16*)ws;
  _Float16* Wk16 = Wq16 + WSZ;
  _Float16* Wv16 = Wk16 + WSZ;
  _Float16* Wo16 = Wv16 + WSZ;
  _Float16* Qh = Wo16 + WSZ;   // [B,H,S,hd]
  _Float16* Kh = Qh + TSZ;     // [B,H,S,hd]
  _Float16* Vt = Kh + TSZ;     // [B,H,hd,S]
  _Float16* Oh = Vt + TSZ;     // [B,S,D]
  // total ws usage: 8 MB weights + 64 MB activations = 72 MB

  cvt_f32_to_f16<<<2048, 256, 0, stream>>>(Wq, Wq16, (int)WSZ);
  cvt_f32_to_f16<<<2048, 256, 0, stream>>>(Wk, Wk16, (int)WSZ);
  cvt_f32_to_f16<<<2048, 256, 0, stream>>>(Wv, Wv16, (int)WSZ);
  cvt_f32_to_f16<<<2048, 256, 0, stream>>>(Wo, Wo16, (int)WSZ);

  dim3 gg(D_MODEL / 128, M_TOK / 128);  // (8, 64)
  gemm_kernel<0><<<gg, 256, 0, stream>>>((const void*)query, Wq16, bq, (void*)Qh);
  gemm_kernel<0><<<gg, 256, 0, stream>>>((const void*)key,   Wk16, bk, (void*)Kh);
  gemm_kernel<1><<<gg, 256, 0, stream>>>((const void*)value, Wv16, bv, (void*)Vt);

  dim3 ag(SEQ / 64, BATCH * NUM_HEADS);  // (32, 64)
  attn_kernel<<<ag, 128, 0, stream>>>(Qh, Kh, Vt, Oh);

  gemm_kernel<2><<<gg, 256, 0, stream>>>((const void*)Oh, Wo16, bo, (void*)out);
}
